// GATNet_simple_withMLP_44281112822532
// MI455X (gfx1250) — matrix operand order repaired and checked by reading, not compiled, not run on hardware
//
#include <hip/hip_runtime.h>
#include <hip/hip_bf16.h>
#include <math.h>

typedef __attribute__((ext_vector_type(16))) _Float16 v16h;
typedef __attribute__((ext_vector_type(8)))  float    v8f;

// =========================================================================
// Fragment packing (ISA 7.12.2 16-bit layouts), f32 -> f16.
//
// A (16x32 MxK tile): lane = (m&15) + ((k%16)>=8 ? 16 : 0)
//                     j    = (k>=16 ? 8 : 0) + (k&7)
// B (32x16 KxN tile): lane = (n&15) + ((k%16)>=8 ? 16 : 0)
//                     j    = (k>=16 ? 8 : 0) + (k&7)
// Packed layout: [tile][kblock][lane][j], 512 halves per (tile,kblock).
// =========================================================================

__global__ void pack_a(const float* __restrict__ A, _Float16* __restrict__ Ap,
                       int M, int K, int Kpad) {
    int i = blockIdx.x * blockDim.x + threadIdx.x;
    if (i >= M * Kpad) return;
    int m = i / Kpad, k = i % Kpad;
    float v = (k < K) ? A[(size_t)m * K + k] : 0.0f;
    int mt = m >> 4, mr = m & 15;
    int kb = k >> 5, kk = k & 31;
    int lane = mr + (((kk & 15) >= 8) ? 16 : 0);
    int j    = ((kk >= 16) ? 8 : 0) + (kk & 7);
    Ap[((size_t)mt * (Kpad >> 5) + kb) * 512 + lane * 16 + j] = (_Float16)v;
}

__global__ void pack_b(const float* __restrict__ B, _Float16* __restrict__ Bp,
                       int K, int Kpad, int N) {
    int i = blockIdx.x * blockDim.x + threadIdx.x;
    if (i >= Kpad * N) return;
    int k = i / N, n = i % N;                 // coalesced read over n
    float v = (k < K) ? B[(size_t)k * N + n] : 0.0f;
    int nt = n >> 4, nr = n & 15;
    int kb = k >> 5, kk = k & 31;
    int lane = nr + (((kk & 15) >= 8) ? 16 : 0);
    int j    = ((kk >= 16) ? 8 : 0) + (kk & 7);
    Bp[((size_t)nt * (Kpad >> 5) + kb) * 512 + lane * 16 + j] = (_Float16)v;
}

// =========================================================================
// C[M,N] = A @ B from fragment-packed f16 operands. One wave per 16x16 C
// tile; block = 4 waves = 4 consecutive n-tiles of one m-tile. Inner loop:
// 2x v16h global loads (4x b128) + v_wmma_f32_16x16x32_f16. No LDS.
// Requires M%16==0, N%64==0 (true for all call sites).
// =========================================================================
__global__ void gemm_wmma_packed(const _Float16* __restrict__ Ap,
                                 const _Float16* __restrict__ Bp,
                                 float* __restrict__ C,
                                 int M, int Kblocks, int N) {
    const int lane = threadIdx.x & 31;
    const int wv   = threadIdx.x >> 5;
    const int nt   = blockIdx.x * 4 + wv;     // n-tile (wave-uniform)
    const int mt   = blockIdx.y;              // m-tile
    if (nt * 16 >= N) return;                 // wave-uniform guard

    const _Float16* ap = Ap + (size_t)mt * Kblocks * 512 + lane * 16;
    const _Float16* bp = Bp + (size_t)nt * Kblocks * 512 + lane * 16;

    v8f acc = {};
    #pragma unroll 2
    for (int kb = 0; kb < Kblocks; ++kb) {
        v16h a = *(const v16h*)ap;  ap += 512;
        v16h b = *(const v16h*)bp;  bp += 512;
        acc = __builtin_amdgcn_wmma_f32_16x16x32_f16(
                  false, a, false, b, (short)0, acc, false, false);
    }

    // C/D layout: VGPR r: lanes 0-15 -> M=r, lanes 16-31 -> M=8+r; N=lane%16
    const int mo = (lane < 16) ? 0 : 8;
    const int nn = nt * 16 + (lane & 15);
    #pragma unroll
    for (int r = 0; r < 8; ++r) {
        C[(size_t)(mt * 16 + mo + r) * N + nn] = acc[r];
    }
}

// -------------------------------------------------------------------------
// attention scores: al[n,h] = sum_c h[n,h,c]*a_s[h,c] ; ar likewise
// -------------------------------------------------------------------------
__global__ void att_scores(const float* __restrict__ h, const float* __restrict__ a_s,
                           const float* __restrict__ a_d, float* __restrict__ al,
                           float* __restrict__ ar, int Nn, int H, int C) {
    int i = blockIdx.x * blockDim.x + threadIdx.x;
    if (i >= Nn * H) return;
    int n = i / H, hh = i % H;
    const float* hp = h + ((size_t)n * H + hh) * C;
    const float* as = a_s + hh * C;
    const float* ad = a_d + hh * C;
    float sl = 0.f, sr = 0.f;
    for (int c = 0; c < C; ++c) { float v = hp[c]; sl += v * as[c]; sr += v * ad[c]; }
    al[i] = sl; ar[i] = sr;
}

__global__ void fill_f32(float* p, float v, int n) {
    int i = blockIdx.x * blockDim.x + threadIdx.x;
    if (i < n) p[i] = v;
}

__device__ __forceinline__ int edge_src(const int* ei, int e, int E) {
    return (e < E) ? ei[e] : (e - E);
}
__device__ __forceinline__ int edge_dst(const int* ei, int e, int E) {
    return (e < E) ? ei[E + e] : (e - E);
}

// e = leaky_relu(al[src]+ar[dst], 0.2); atomic segment-max into m[dst,h] (bit trick)
__global__ void edge_max(const int* __restrict__ ei, const float* __restrict__ al,
                         const float* __restrict__ ar, float* __restrict__ ebuf,
                         int* __restrict__ mbits, int Etot, int E, int H) {
    int i = blockIdx.x * blockDim.x + threadIdx.x;
    if (i >= Etot * H) return;
    int e = i / H, hh = i % H;
    int s = edge_src(ei, e, E), d = edge_dst(ei, e, E);
    float v = al[s * H + hh] + ar[d * H + hh];
    v = (v > 0.f) ? v : 0.2f * v;
    ebuf[i] = v;
    int* addr = mbits + d * H + hh;
    int iv = __float_as_int(v);
    if (iv >= 0) atomicMax(addr, iv);
    else         atomicMin((unsigned int*)addr, (unsigned int)iv);
}

// ex = exp(e - m[dst]); atomic segment-sum into ssum[dst,h]
__global__ void edge_expsum(const int* __restrict__ ei, const float* __restrict__ m,
                            float* __restrict__ ebuf, float* __restrict__ ssum,
                            int Etot, int E, int H) {
    int i = blockIdx.x * blockDim.x + threadIdx.x;
    if (i >= Etot * H) return;
    int e = i / H, hh = i % H;
    int d = edge_dst(ei, e, E);
    float ex = __expf(ebuf[i] - m[d * H + hh]);
    ebuf[i] = ex;
    atomicAdd(&ssum[d * H + hh], ex);
}

// out[dst,h,:] += (ex/ssum[dst,h]) * h[src,h,:]
__global__ void edge_scatter(const int* __restrict__ ei, const float* __restrict__ h,
                             const float* __restrict__ ebuf, const float* __restrict__ ssum,
                             float* __restrict__ out, int Etot, int E, int H, int C) {
    int i = blockIdx.x * blockDim.x + threadIdx.x;
    if (i >= Etot * H) return;
    int e = i / H, hh = i % H;
    int s = edge_src(ei, e, E), d = edge_dst(ei, e, E);
    float alpha = ebuf[i] / ssum[d * H + hh];
    const float* hp = h   + ((size_t)s * H + hh) * C;
    float*       op = out + ((size_t)d * H + hh) * C;
    for (int c = 0; c < C; ++c) atomicAdd(&op[c], alpha * hp[c]);
}

__global__ void bias_relu(const float* __restrict__ acc, const float* __restrict__ b,
                          float* __restrict__ xout, int Nn, int F) {
    int i = blockIdx.x * blockDim.x + threadIdx.x;
    if (i >= Nn * F) return;
    float v = acc[i] + b[i % F];
    xout[i] = (v > 0.f) ? v : 0.f;
}

// out1 = sigmoid([x1|x2|x3] @ Wf + bf) ; one wave per node
__global__ void final_linear(const float* __restrict__ x1, const float* __restrict__ x2,
                             const float* __restrict__ x3, const float* __restrict__ Wf,
                             const float* __restrict__ bf, float* __restrict__ out1, int Nn) {
    int node = blockIdx.x * (blockDim.x >> 5) + (threadIdx.x >> 5);
    int lane = threadIdx.x & 31;
    if (node >= Nn) return;
    float s = 0.f;
    for (int j = lane; j < 256; j += 32) s += x1[(size_t)node * 256 + j] * Wf[j];
    for (int j = lane; j < 256; j += 32) s += x2[(size_t)node * 256 + j] * Wf[256 + j];
    for (int j = lane; j < 768; j += 32) s += x3[(size_t)node * 768 + j] * Wf[512 + j];
    for (int off = 16; off > 0; off >>= 1) s += __shfl_down(s, off, 32);
    if (lane == 0) out1[node] = 1.f / (1.f + __expf(-(s + bf[0])));
}

__global__ void build_xm(const float* __restrict__ x, const float* __restrict__ out1,
                         float* __restrict__ xm, int Nn) {
    int i = blockIdx.x * blockDim.x + threadIdx.x;
    if (i >= Nn * 17) return;
    int n = i / 17, c = i % 17;
    xm[i] = (c < 16) ? x[n * 16 + c] : out1[n];
}

// out2 = sigmoid(mh2[node,0:64] . M3w + M3b) ; one wave per node
__global__ void mlp_out(const float* __restrict__ mh2, const float* __restrict__ M3w,
                        const float* __restrict__ M3b, float* __restrict__ out2, int Nn) {
    int node = blockIdx.x * (blockDim.x >> 5) + (threadIdx.x >> 5);
    int lane = threadIdx.x & 31;
    if (node >= Nn) return;
    float s = mh2[(size_t)node * 64 + lane] * M3w[lane]
            + mh2[(size_t)node * 64 + 32 + lane] * M3w[32 + lane];
    for (int off = 16; off > 0; off >>= 1) s += __shfl_down(s, off, 32);
    if (lane == 0) out2[node] = 1.f / (1.f + __expf(-(s + M3b[0])));
}

// =========================================================================

extern "C" void kernel_launch(void* const* d_in, const int* in_sizes, int n_in,
                              void* d_out, int out_size, void* d_ws, size_t ws_size,
                              hipStream_t stream) {
    const int N    = in_sizes[0] / 16;
    const int E    = in_sizes[1] / 2;
    const int Etot = E + N;

    const float* x   = (const float*)d_in[0];
    const int*   ei  = (const int*)  d_in[1];
    // d_in[2] = JetRawPt (unused by the reference network)
    const float* W1  = (const float*)d_in[3];
    const float* a1s = (const float*)d_in[4];
    const float* a1d = (const float*)d_in[5];
    const float* b1  = (const float*)d_in[6];
    const float* W2  = (const float*)d_in[7];
    const float* a2s = (const float*)d_in[8];
    const float* a2d = (const float*)d_in[9];
    const float* b2  = (const float*)d_in[10];
    const float* W3  = (const float*)d_in[11];
    const float* a3s = (const float*)d_in[12];
    const float* a3d = (const float*)d_in[13];
    const float* b3  = (const float*)d_in[14];
    const float* Wf  = (const float*)d_in[15];
    const float* bf  = (const float*)d_in[16];
    const float* M1w = (const float*)d_in[17];
    const float* M1b = (const float*)d_in[18];
    const float* M2w = (const float*)d_in[19];
    const float* M2b = (const float*)d_in[20];
    const float* M3w = (const float*)d_in[21];
    const float* M3b = (const float*)d_in[22];

    float* out1 = (float*)d_out;
    float* out2 = (float*)d_out + N;

    // workspace carve-up (256B aligned)
    size_t off = 0;
    auto carveB = [&](size_t bytes) {
        void* p = (char*)d_ws + off;
        off += (bytes + 255) & ~(size_t)255;
        return p;
    };
    auto carve  = [&](size_t cnt) { return (float*)carveB(cnt * sizeof(float)); };

    float*    x1     = carve((size_t)N * 256);
    float*    x2     = carve((size_t)N * 256);
    float*    x3     = carve((size_t)N * 768);
    float*    hbuf   = carve((size_t)N * 768);            // reused each layer
    float*    accbuf = carve((size_t)N * 768);            // reused each layer + MLP temp
    float*    al     = carve((size_t)N * 12);
    float*    ar     = carve((size_t)N * 12);
    float*    mmax   = carve((size_t)N * 12);
    float*    ssum   = carve((size_t)N * 12);
    float*    ebuf   = carve((size_t)Etot * 12);
    float*    xm     = carve((size_t)N * 17);
    float*    t128   = carve((size_t)N * 128);
    float*    t64    = carve((size_t)N * 64);
    _Float16* apack  = (_Float16*)carveB((size_t)N * 256 * sizeof(_Float16));
    _Float16* bpack  = (_Float16*)carveB((size_t)768 * 256 * sizeof(_Float16));
    (void)ws_size; (void)n_in; (void)out_size;

    auto cdiv = [](int a, int b) { return (a + b - 1) / b; };

    // GEMM: Cout[M,Nc] = Ain[M,K] @ W[K,Nc]   (pack -> WMMA)
    auto run_gemm = [&](const float* Ain, const float* W, float* Cout,
                        int M, int K, int Nc) {
        const int Kpad = (K + 31) & ~31;
        const int Kb   = Kpad >> 5;
        pack_a<<<cdiv(M * Kpad, 256), 256, 0, stream>>>(Ain, apack, M, K, Kpad);
        pack_b<<<cdiv(Kpad * Nc, 256), 256, 0, stream>>>(W, bpack, K, Kpad, Nc);
        gemm_wmma_packed<<<dim3(Nc / 64, M / 16), 128, 0, stream>>>(apack, bpack, Cout, M, Kb, Nc);
    };

    auto gat_layer = [&](const float* xin, int Fin, const float* W, const float* as_,
                         const float* ad_, const float* b, int H, int C, float* xout) {
        const int F = H * C;
        run_gemm(xin, W, hbuf, N, Fin, F);                       // h = xin @ W
        att_scores<<<cdiv(N * H, 256), 256, 0, stream>>>(hbuf, as_, ad_, al, ar, N, H, C);
        fill_f32<<<cdiv(N * H, 256), 256, 0, stream>>>(mmax, -INFINITY, N * H);
        hipMemsetAsync(ssum, 0, (size_t)N * H * sizeof(float), stream);
        hipMemsetAsync(accbuf, 0, (size_t)N * F * sizeof(float), stream);
        int EH = Etot * H;
        edge_max    <<<cdiv(EH, 256), 256, 0, stream>>>(ei, al, ar, ebuf, (int*)mmax, Etot, E, H);
        edge_expsum <<<cdiv(EH, 256), 256, 0, stream>>>(ei, mmax, ebuf, ssum, Etot, E, H);
        edge_scatter<<<cdiv(EH, 256), 256, 0, stream>>>(ei, hbuf, ebuf, ssum, accbuf, Etot, E, H, C);
        bias_relu   <<<cdiv(N * F, 256), 256, 0, stream>>>(accbuf, b, xout, N, F);
    };

    gat_layer(x,  16,  W1, a1s, a1d, b1,  8, 32, x1);
    gat_layer(x1, 256, W2, a2s, a2d, b2,  8, 32, x2);
    gat_layer(x2, 256, W3, a3s, a3d, b3, 12, 64, x3);

    // out1 = sigmoid(concat(x1,x2,x3) @ Wf + bf)
    final_linear<<<cdiv(N, 8), 256, 0, stream>>>(x1, x2, x3, Wf, bf, out1, N);

    // MLP on [x | out1]
    build_xm<<<cdiv(N * 17, 256), 256, 0, stream>>>(x, out1, xm, N);
    run_gemm(xm,   M1w, accbuf, N, 17,  128);
    bias_relu<<<cdiv(N * 128, 256), 256, 0, stream>>>(accbuf, M1b, t128, N, 128);
    run_gemm(t128, M2w, accbuf, N, 128, 64);
    bias_relu<<<cdiv(N * 64, 256), 256, 0, stream>>>(accbuf, M2b, t64, N, 64);
    mlp_out<<<cdiv(N, 8), 256, 0, stream>>>(t64, M3w, M3b, out2, N);
}
